// MultiheadAttention_72748156060162
// MI455X (gfx1250) — compile-verified, matrix operand order checked by reading, not run on hardware
//
#include <hip/hip_runtime.h>
#include <hip/hip_bf16.h>
#include <stdint.h>

// MI455X / gfx1250 multihead attention:
//   K1: QKV projection GEMMs (f32 in -> f16 out, WMMA f16, fp32 accum)
//   K2: flash-attention (online softmax, WMMA f16, TDM async K/Q tile loads,
//       double-buffered K chunks)
//   K3: output projection GEMM (f16 A -> f32 out)

typedef __attribute__((ext_vector_type(16))) _Float16 v16h;
typedef __attribute__((ext_vector_type(8)))  float    v8f;
typedef __attribute__((ext_vector_type(4)))  uint32_t v4u;
typedef __attribute__((ext_vector_type(8)))  int      v8i_;
typedef __attribute__((ext_vector_type(4)))  int      v4i_;

#define B_    4
#define T_    2048
#define EMB_  512
#define H_    8
#define HK_   4096              // H_*EMB_
#define MROWS_ (B_*T_)          // 8192
#define NCH_  (T_/32)           // 64 KV chunks
#define QKSCALE 0.21022410381342865f  // 512^(-1/4)

#if defined(__has_builtin)
#  if __has_builtin(__builtin_amdgcn_tensor_load_to_lds) && \
      __has_builtin(__builtin_amdgcn_s_wait_tensorcnt)
#    define HAVE_TDM 1
#  endif
#endif
#ifndef HAVE_TDM
#  define HAVE_TDM 0
#endif

union FragU   { v16h v; _Float16 h[16]; uint32_t u[8]; };
union HalfPair{ uint32_t u; _Float16 h[2]; };

// Load a 16x32 f16 fragment (A or B operand) from a tile stored so that the
// 32-wide K dimension is contiguous per row (row = M for A, row = N for B).
// CDNA5 16-bit layout: lanes 0-15 rows 0-15 / K-half 0, lanes 16-31 K-half 1;
// dword j<4 -> K=2j+8*kh, j>=4 -> K=16+2(j-4)+8*kh.
__device__ __forceinline__ v16h load_frag16x32(const _Float16* base, int strideH, int lane) {
  FragU f;
  const int r  = lane & 15;
  const int kh = (lane >> 4) & 1;
  const uint32_t* rp = (const uint32_t*)(base + (size_t)r * strideH);
#pragma unroll
  for (int j = 0; j < 4; ++j) f.u[j] = rp[j + 4 * kh];
#pragma unroll
  for (int j = 0; j < 4; ++j) f.u[4 + j] = rp[8 + j + 4 * kh];
  return f.v;
}

__device__ __forceinline__ v8f wmma16(v16h a, v16h b, v8f c) {
  return __builtin_amdgcn_wmma_f32_16x16x32_f16(false, a, false, b, (short)0, c, false, false);
}

__device__ __forceinline__ float grp16_max(float x) {
#pragma unroll
  for (int m = 1; m < 16; m <<= 1) x = fmaxf(x, __shfl_xor(x, m, 32));
  return x;
}
__device__ __forceinline__ float grp16_sum(float x) {
#pragma unroll
  for (int m = 1; m < 16; m <<= 1) x += __shfl_xor(x, m, 32);
  return x;
}

#if HAVE_TDM
// Issue a 2D TENSOR_LOAD_TO_LDS: rows x width f16 tile, row-major in LDS,
// source row stride = row_stride_elems (f16 units). D# per ISA 08 §8.3/8.4.
__device__ __forceinline__ void tdm_load_2d(uint32_t lds_byte_addr, const void* gptr,
                                            uint32_t width_elems, uint32_t rows,
                                            uint32_t row_stride_elems) {
  const uint64_t ga = (uint64_t)(uintptr_t)gptr;
  v4u g0;
  g0[0] = 1u;                                       // count=1, user descriptor
  g0[1] = lds_byte_addr;                            // lds_addr
  g0[2] = (uint32_t)ga;                             // global_addr[31:0]
  g0[3] = (uint32_t)((ga >> 32) & 0x01FFFFFFu)      // global_addr[56:32]
          | (2u << 30);                             // type = 2 (image)
  v8i_ g1;
  g1[0] = (int)(1u << 16);                          // wg_mask=0, data_size=1 (2B)
  g1[1] = (int)((width_elems & 0xFFFFu) << 16);     // tensor_dim0[15:0]
  g1[2] = (int)(((width_elems >> 16) & 0xFFFFu)     // tensor_dim0[31:16]
          | ((rows & 0xFFFFu) << 16));              // tensor_dim1[15:0]
  g1[3] = (int)(((rows >> 16) & 0xFFFFu)            // tensor_dim1[31:16]
          | ((width_elems & 0xFFFFu) << 16));       // tile_dim0
  g1[4] = (int)(rows & 0xFFFFu);                    // tile_dim1 (tile_dim2=0)
  g1[5] = (int)row_stride_elems;                    // tensor_dim0_stride[31:0]
  g1[6] = 0;                                        // stride hi, dim1_stride lo
  g1[7] = 0;
  const v4i_ z4 = {0, 0, 0, 0};
#if __clang_major__ >= 23
  const v8i_ z8 = {0, 0, 0, 0, 0, 0, 0, 0};
  __builtin_amdgcn_tensor_load_to_lds(g0, g1, z4, z4, z8, 0);
#else
  __builtin_amdgcn_tensor_load_to_lds(g0, g1, z4, z4, 0);
#endif
}
__device__ __forceinline__ uint32_t lds_addr_of(const void* p) {
  return (uint32_t)(uintptr_t)p;   // flat shared-aperture low bits = LDS byte addr
}
#endif

// ---------------------------------------------------------------------------
// Tiled WMMA GEMM: C[M,N] = (A[M,K] @ W[K,N] + bias) * scale
// Block tile 128(M) x 64(N), K-step 32. 8 waves: wave w owns rows w*16..w*16+15.
// ---------------------------------------------------------------------------
template <typename AT, typename OT>
__global__ __launch_bounds__(256) void gemm_wmma_k(
    const AT* __restrict__ A, int lda,
    const float* __restrict__ W, int ldw,
    const float* __restrict__ bias,
    OT* __restrict__ C, int ldc,
    int Kdim, float scale)
{
  __shared__ _Float16 as_[128 * 32];   // A tile, row-major (K contiguous)
  __shared__ _Float16 bs_[64 * 32];    // W tile transposed: bs_[n*32 + k]

  const int tid  = threadIdx.x;
  const int lane = tid & 31;
  const int wv   = tid >> 5;
  const int mbase = blockIdx.y * 128;
  const int nbase = blockIdx.x * 64;

  v8f acc[4];
#pragma unroll
  for (int nt = 0; nt < 4; ++nt)
    acc[nt] = (v8f){0.f, 0.f, 0.f, 0.f, 0.f, 0.f, 0.f, 0.f};

  for (int kt = 0; kt < Kdim; kt += 32) {
    __syncthreads();
#pragma unroll
    for (int e = 0; e < 16; ++e) {            // 128*32 = 4096 elems / 256 thr
      int idx = e * 256 + tid;
      int r = idx >> 5, k = idx & 31;
      as_[idx] = (_Float16)A[(size_t)(mbase + r) * lda + kt + k];
    }
#pragma unroll
    for (int e = 0; e < 8; ++e) {             // 32*64 = 2048 elems
      int idx = e * 256 + tid;
      int k = idx >> 6, n = idx & 63;
      bs_[n * 32 + k] = (_Float16)W[(size_t)(kt + k) * ldw + nbase + n];
    }
    __syncthreads();

    v16h af = load_frag16x32(as_ + wv * 16 * 32, 32, lane);
#pragma unroll
    for (int nt = 0; nt < 4; ++nt) {
      v16h bf = load_frag16x32(bs_ + nt * 16 * 32, 32, lane);
      acc[nt] = wmma16(af, bf, acc[nt]);
    }
  }

  const int g = lane >> 4, c0 = lane & 15;
#pragma unroll
  for (int nt = 0; nt < 4; ++nt) {
#pragma unroll
    for (int i = 0; i < 8; ++i) {
      int r = mbase + wv * 16 + i + 8 * g;
      int c = nbase + nt * 16 + c0;
      float v = (acc[nt][i] + bias[c]) * scale;
      C[(size_t)r * ldc + c] = (OT)v;
    }
  }
}

// ---------------------------------------------------------------------------
// Flash attention over one (b, h, 64-query block).
// Scores phase: wave = (qt = w&3, kt = w>>2) -> one 16x16 S tile.
// PV phase:     wave = (qt = w&3, d-half = w>>2) -> 16x256 f32 accumulator.
// Q block + K chunks loaded by the Tensor Data Mover (double-buffered K).
// ---------------------------------------------------------------------------
__global__ __launch_bounds__(256) void flash_attn_k(
    const _Float16* __restrict__ Q,
    const _Float16* __restrict__ K,
    const _Float16* __restrict__ V,
    _Float16* __restrict__ O)
{
  extern __shared__ char smem[];
  _Float16* qs  = (_Float16*)smem;      // 64 x 512 (row-major, d contiguous)
  _Float16* ks0 = qs  + 64 * 512;       // 32 x 512 K chunk, buffer 0
  _Float16* ks1 = ks0 + 32 * 512;       // 32 x 512 K chunk, buffer 1
  _Float16* vt  = ks1 + 32 * 512;       // 512 x 32 transposed: vt[d*32 + kv]
  _Float16* ps  = vt  + 512 * 32;       // 64 x 32 P tile (kv contiguous)
  float* m_arr = (float*)(ps + 64 * 32);  // [64] running row max
  float* l_arr = m_arr + 64;              // [64] running row sum
  float* al    = l_arr + 64;              // [64] alpha = exp(m_old - m_new)
  float* mpart = al + 64;                 // [2][64] per-kt partial max
  float* spart = mpart + 128;             // [2][64] per-kt partial sum

  const int tid  = threadIdx.x;
  const int lane = tid & 31;
  const int wv   = tid >> 5;
  const int qt   = wv & 3;     // query sub-tile 0..3
  const int kt   = wv >> 2;    // key sub-tile / d-half 0..1
  const int g    = lane >> 4;
  const int c0   = lane & 15;

  const int qblk = blockIdx.x, h = blockIdx.y, b = blockIdx.z;
  const size_t qoff = (size_t)(b * T_ + qblk * 64) * HK_ + (size_t)h * EMB_;
  const size_t kv0  = (size_t)(b * T_) * HK_ + (size_t)h * EMB_;

#if HAVE_TDM
  // Wave 0 issues DMA for Q block and the first K chunk; DMA runs while the
  // workgroup initializes state / loads V.
  if (wv == 0) {
    tdm_load_2d(lds_addr_of(qs),  Q + qoff, EMB_, 64, HK_);
    tdm_load_2d(lds_addr_of(ks0), K + kv0,  EMB_, 32, HK_);
  }
#else
#pragma unroll 4
  for (int e = 0; e < 64; ++e) {         // 64 x 256 dwords
    int idx = e * 256 + tid;
    int r = idx >> 8, c2 = idx & 255;
    ((uint32_t*)qs)[idx] = *(const uint32_t*)(Q + qoff + (size_t)r * HK_ + c2 * 2);
  }
#pragma unroll 4
  for (int e = 0; e < 32; ++e) {         // first K chunk, 32 x 256 dwords
    int idx = e * 256 + tid;
    int r = idx >> 8, c2 = idx & 255;
    ((uint32_t*)ks0)[idx] = *(const uint32_t*)(K + kv0 + (size_t)r * HK_ + c2 * 2);
  }
#endif
  if (tid < 64) { m_arr[tid] = -3.0e38f; l_arr[tid] = 0.f; }

  v8f oacc[16];
#pragma unroll
  for (int nt = 0; nt < 16; ++nt)
    oacc[nt] = (v8f){0.f, 0.f, 0.f, 0.f, 0.f, 0.f, 0.f, 0.f};

  for (int ch = 0; ch < NCH_; ++ch) {
    __syncthreads();   // all compute on previous chunk's ks/vt/ps done
    const size_t kvoff = kv0 + (size_t)(ch * 32) * HK_;
    _Float16* ksb = (ch & 1) ? ks1 : ks0;

#if HAVE_TDM
    if (wv == 0 && ch + 1 < NCH_) {      // prefetch next K chunk via TDM
      _Float16* ksn = ((ch + 1) & 1) ? ks1 : ks0;
      tdm_load_2d(lds_addr_of(ksn), K + kvoff + (size_t)32 * HK_, EMB_, 32, HK_);
    }
#endif
    // V chunk: load + transpose into vt[d][kv] (makes PV B-fragments contiguous)
#pragma unroll 4
    for (int e = 0; e < 32; ++e) {       // 32 rows x 256 dwords
      int idx = e * 256 + tid;
      int r = idx >> 8, c2 = idx & 255;
      HalfPair hp;
      hp.u = *(const uint32_t*)(V + kvoff + (size_t)r * HK_ + c2 * 2);
      vt[(2 * c2) * 32 + r]     = hp.h[0];
      vt[(2 * c2 + 1) * 32 + r] = hp.h[1];
#if !HAVE_TDM
      if (ch + 1 < NCH_)
        ((uint32_t*)((ch & 1) ? ks0 : ks1))[idx] =
            *(const uint32_t*)(K + kvoff + (size_t)(32 + r) * HK_ + c2 * 2);
#endif
    }
    if (ch + 1 < NCH_)                   // -> global_prefetch_b8
      __builtin_prefetch(V + kvoff + (size_t)32 * HK_ + tid * 128, 0, 1);
#if HAVE_TDM
    if (wv == 0) {                       // newest descriptor may stay in flight
      if (ch + 1 < NCH_) __builtin_amdgcn_s_wait_tensorcnt(1);
      else               __builtin_amdgcn_s_wait_tensorcnt(0);
    }
#endif
    __syncthreads();

#if !HAVE_TDM
    if (ch == 0) { /* ks0 already loaded pre-loop; ks1 filled above for ch=1 */ }
#endif

    // ---- scores: S(16x16) = Q[qt] . K[kt]^T, summed over d=512 ----
    v8f s = (v8f){0.f, 0.f, 0.f, 0.f, 0.f, 0.f, 0.f, 0.f};
#pragma unroll
    for (int kk = 0; kk < 512; kk += 32) {
      v16h aq = load_frag16x32(qs  + qt * 16 * 512 + kk, 512, lane);
      v16h bk = load_frag16x32(ksb + kt * 16 * 512 + kk, 512, lane);
      s = wmma16(aq, bk, s);
    }

    // ---- online softmax: per-row max over this chunk ----
    float rm[8];
#pragma unroll
    for (int i = 0; i < 8; ++i) rm[i] = grp16_max(s[i]);
    if (c0 == 0) {
#pragma unroll
      for (int i = 0; i < 8; ++i)
        mpart[kt * 64 + qt * 16 + i + 8 * g] = rm[i];
    }
    __syncthreads();
    if (tid < 64) {
      float nm = fmaxf(m_arr[tid], fmaxf(mpart[tid], mpart[64 + tid]));
      al[tid] = __expf(m_arr[tid] - nm);
      m_arr[tid] = nm;
    }
    __syncthreads();

    // rescale running O by alpha
    float av[8];
#pragma unroll
    for (int i = 0; i < 8; ++i) av[i] = al[qt * 16 + i + 8 * g];
#pragma unroll
    for (int nt = 0; nt < 16; ++nt)
#pragma unroll
      for (int i = 0; i < 8; ++i) oacc[nt][i] *= av[i];

    // P = exp(S - m_new), write f16 P tile + partial row sums
    float mr[8], rs[8];
#pragma unroll
    for (int i = 0; i < 8; ++i) mr[i] = m_arr[qt * 16 + i + 8 * g];
#pragma unroll
    for (int i = 0; i < 8; ++i) {
      float p = __expf(s[i] - mr[i]);
      ps[(qt * 16 + i + 8 * g) * 32 + kt * 16 + c0] = (_Float16)p;
      rs[i] = grp16_sum(p);
    }
    if (c0 == 0) {
#pragma unroll
      for (int i = 0; i < 8; ++i)
        spart[kt * 64 + qt * 16 + i + 8 * g] = rs[i];
    }
    __syncthreads();
    if (tid < 64)
      l_arr[tid] = l_arr[tid] * al[tid] + spart[tid] + spart[64 + tid];

    // ---- PV: O[qt, d-half] += P(16x32) . V(32x256) ----
    v16h ap = load_frag16x32(ps + qt * 16 * 32, 32, lane);
#pragma unroll
    for (int nt = 0; nt < 16; ++nt) {
      v16h bv = load_frag16x32(vt + (kt * 256 + nt * 16) * 32, 32, lane);
      oacc[nt] = wmma16(ap, bv, oacc[nt]);
    }
  }

  __syncthreads();
  float rl[8];
#pragma unroll
  for (int i = 0; i < 8; ++i) rl[i] = 1.f / l_arr[qt * 16 + i + 8 * g];
#pragma unroll
  for (int nt = 0; nt < 16; ++nt) {
#pragma unroll
    for (int i = 0; i < 8; ++i) {
      int r = qblk * 64 + qt * 16 + i + 8 * g;
      int d = h * EMB_ + kt * 256 + nt * 16 + c0;
      O[(size_t)(b * T_ + r) * HK_ + d] = (_Float16)(oacc[nt][i] * rl[i]);
    }
  }
}

// ---------------------------------------------------------------------------
extern "C" void kernel_launch(void* const* d_in, const int* in_sizes, int n_in,
                              void* d_out, int out_size, void* d_ws, size_t ws_size,
                              hipStream_t stream) {
  (void)in_sizes; (void)n_in; (void)out_size; (void)ws_size;
  const float* x  = (const float*)d_in[0];
  const float* Wq = (const float*)d_in[1];
  const float* bq = (const float*)d_in[2];
  const float* Wk = (const float*)d_in[3];
  const float* bk = (const float*)d_in[4];
  const float* Wv = (const float*)d_in[5];
  const float* bv = (const float*)d_in[6];
  const float* Wo = (const float*)d_in[7];
  const float* bo = (const float*)d_in[8];
  float* out = (float*)d_out;

  // Workspace: Q, K, V, attn-out in f16, each MROWS_ x HK_ (64 MB) = 256 MB.
  _Float16* qh = (_Float16*)d_ws;
  _Float16* kh = qh + (size_t)MROWS_ * HK_;
  _Float16* vh = kh + (size_t)MROWS_ * HK_;
  _Float16* ah = vh + (size_t)MROWS_ * HK_;

  dim3 pgrid(HK_ / 64, MROWS_ / 128);   // 64 x 64
  gemm_wmma_k<float, _Float16><<<pgrid, 256, 0, stream>>>(
      x, EMB_, Wq, HK_, bq, qh, HK_, EMB_, QKSCALE);
  gemm_wmma_k<float, _Float16><<<pgrid, 256, 0, stream>>>(
      x, EMB_, Wk, HK_, bk, kh, HK_, EMB_, QKSCALE);
  gemm_wmma_k<float, _Float16><<<pgrid, 256, 0, stream>>>(
      x, EMB_, Wv, HK_, bv, vh, HK_, EMB_, 1.0f);

  const size_t smem_bytes =
      ((size_t)64 * 512 + 2 * 32 * 512 + 512 * 32 + 64 * 32) * sizeof(_Float16) +
      (64 * 3 + 128 * 2) * sizeof(float);    // 169728 B (< 320 KB WGP LDS)
  flash_attn_k<<<dim3(T_ / 64, H_, B_), 256, smem_bytes, stream>>>(qh, kh, vh, ah);

  dim3 ogrid(EMB_ / 64, MROWS_ / 128);  // 8 x 64
  gemm_wmma_k<_Float16, float><<<ogrid, 256, 0, stream>>>(
      ah, HK_, Wo, EMB_, bo, out, EMB_, HK_, 1.0f);
}
